// ISA_Block_10307921511200
// MI455X (gfx1250) — compile-verified, matrix operand order checked by reading, not run on hardware
//
#include <hip/hip_runtime.h>

// ---------------------------------------------------------------------------
// ISA-Block two-stage attention for MI455X (gfx1250, wave32, WMMA bf16).
//   C=256, H=8, d=32, B=2, N=16384.  Stage1: L=64, Nb=512.  Stage2: L=256, Nb=128.
// All GEMM/attention math: v_wmma_f32_16x16x32_bf16 (bf16 in, f32 accumulate).
// ---------------------------------------------------------------------------

typedef __attribute__((ext_vector_type(16))) __bf16       v16bf;
typedef __attribute__((ext_vector_type(8)))  float        v8f;
typedef __attribute__((ext_vector_type(4)))  unsigned int u32x4;
typedef unsigned short u16;

union Frag { v16bf v; u32x4 q[2]; };

__device__ __forceinline__ u16 f2bf(float f) {
  unsigned int u = __float_as_uint(f);
  u += 0x7FFFu + ((u >> 16) & 1u);          // round-to-nearest-even
  return (u16)(u >> 16);
}

__device__ __forceinline__ v8f wmma_bf16(const Frag& a, const Frag& b, v8f c) {
  // (neg_a, A, neg_b, B, c_mod, C, reuse_a, reuse_b)
  return __builtin_amdgcn_wmma_f32_16x16x32_bf16(false, a.v, false, b.v,
                                                 (short)0, c, false, false);
}

// ---------------------------------------------------------------------------
// Pack stage-1 inputs into "long" layout [L=64][Nb=512][C] as bf16.
//   xq = q+pos, xk = k+pos, xv = v
// ---------------------------------------------------------------------------
__global__ void pack1_kernel(const float* __restrict__ q, const float* __restrict__ k,
                             const float* __restrict__ v, const float* __restrict__ pos,
                             u16* __restrict__ xq, u16* __restrict__ xk,
                             u16* __restrict__ xv) {
  size_t e = (size_t)blockIdx.x * blockDim.x + threadIdx.x;
  if (e >= (size_t)32768 * 256) return;
  int c  = (int)(e & 255);
  int m  = (int)(e >> 8);
  int nb = m & 511, l = m >> 9;
  int bh = l >> 3, bw = l & 7;                 // bn_h, bn_w
  int b  = nb >> 8, hh = (nb >> 4) & 15, ww = nb & 15;  // B, h_b, w_b
  int n  = ((bh * 16 + hh) * 8 + bw) * 16 + ww;
  size_t src = ((size_t)b * 16384 + n) * 256 + c;
  float p = pos[(size_t)n * 256 + c];
  xq[e] = f2bf(q[src] + p);
  xk[e] = f2bf(k[src] + p);
  xv[e] = f2bf(v[src]);
}

// ---------------------------------------------------------------------------
// f32 -> bf16 weight conversion.
// ---------------------------------------------------------------------------
__global__ void cvt_kernel(const float* __restrict__ s, u16* __restrict__ d, int n) {
  int i = blockIdx.x * blockDim.x + threadIdx.x;
  if (i < n) d[i] = f2bf(s[i]);
}

// ---------------------------------------------------------------------------
// Stage-2 repack: faithful flat-view to_short of v1 plus pos add.
//   short row m2 = l2*128+nb2 with l2=(j2,j4), nb2=(b2,j1,j3);
//   source row in v1 flat [64*512,256]: frow = (((b2*8+j1)*16+j2)*8+j3)*16+j4
// ---------------------------------------------------------------------------
__global__ void pack2_kernel(const float* __restrict__ v1, const float* __restrict__ pos,
                             u16* __restrict__ xqk, u16* __restrict__ xv) {
  size_t e = (size_t)blockIdx.x * blockDim.x + threadIdx.x;
  if (e >= (size_t)32768 * 256) return;
  int c  = (int)(e & 255);
  int m2 = (int)(e >> 8);
  int l2 = m2 >> 7, nb2 = m2 & 127;
  int j2 = l2 >> 4, j4 = l2 & 15;
  int b2 = nb2 >> 6, j1 = (nb2 >> 3) & 7, j3 = nb2 & 7;
  int frow = (((b2 * 8 + j1) * 16 + j2) * 8 + j3) * 16 + j4;
  float val = v1[(size_t)frow * 256 + c];
  // ps = to_short(to_long(pos)): decode frow back through the long layout
  int l1 = frow >> 9, nb1 = frow & 511;
  int bh = l1 >> 3, bw = l1 & 7;
  int hh = (nb1 >> 4) & 15, ww = nb1 & 15;
  int n1 = ((bh * 16 + hh) * 8 + bw) * 16 + ww;
  float p = pos[(size_t)n1 * 256 + c];
  xqk[e] = f2bf(val + p);
  xv[e]  = f2bf(val);
}

// ---------------------------------------------------------------------------
// WMMA GEMM:  out[M,256] = A[M,256](bf16) x W[256,256](bf16, row-major = B^T) + bias
// Block = 128 threads (4 waves); wave computes a 16x64 strip, 4 accumulators.
// A-fragment (16x32 bf16): lane<16 -> K {0..7,16..23}, lane>=16 -> K {8..15,24..31}
//   = two 16-byte chunks per lane.  B fragment = A-layout of B^T = rows of W.
// mode 0: bf16 row-major; 1: f32 row-major; 2: f32 scattered to final [B,N,C].
// ---------------------------------------------------------------------------
__global__ __launch_bounds__(128) void gemm_kernel(const u16* __restrict__ A,
                                                   const u16* __restrict__ W,
                                                   const float* __restrict__ bias,
                                                   void* __restrict__ outp, int mode) {
  const int tid = threadIdx.x, wave = tid >> 5, lane = tid & 31;
  const int hf = lane >> 4, l15 = lane & 15;
  const int m0 = blockIdx.x * 64 + wave * 16;
  const int n0 = blockIdx.y * 64;

  v8f acc[4];
  for (int t = 0; t < 4; ++t)
    for (int r = 0; r < 8; ++r) acc[t][r] = 0.f;

  const int arow = m0 + l15;
  const u16* abase = A + (size_t)arow * 256 + (hf ? 8 : 0);
  for (int kk = 0; kk < 8; ++kk) {
    Frag a;
    const u16* ap = abase + kk * 32;
    __builtin_prefetch(ap + 32, 0, 1);                 // global_prefetch next K-slice
    a.q[0] = *(const u32x4*)ap;
    a.q[1] = *(const u32x4*)(ap + 16);
#pragma unroll
    for (int t = 0; t < 4; ++t) {
      Frag b;
      const u16* wp = W + (size_t)(n0 + t * 16 + l15) * 256 + kk * 32 + (hf ? 8 : 0);
      b.q[0] = *(const u32x4*)wp;
      b.q[1] = *(const u32x4*)(wp + 16);
      acc[t] = wmma_bf16(a, b, acc[t]);
    }
  }

#pragma unroll
  for (int t = 0; t < 4; ++t) {
    int   n  = n0 + t * 16 + l15;
    float bv = bias[n];
#pragma unroll
    for (int r = 0; r < 8; ++r) {
      int   row = m0 + r + hf * 8;                     // C layout: VGPR r -> M=r / r+8
      float v   = acc[t][r] + bv;
      if (mode == 0) {
        ((u16*)outp)[(size_t)row * 256 + n] = f2bf(v);
      } else if (mode == 1) {
        ((float*)outp)[(size_t)row * 256 + n] = v;
      } else {
        int l2 = row >> 7, nb2 = row & 127;
        int j2 = l2 >> 4, j4 = l2 & 15;
        int b2 = nb2 >> 6, j1 = (nb2 >> 3) & 7, j3 = nb2 & 7;
        int nf = ((j2 * 8 + j1) * 16 + j4) * 8 + j3;   // final token index
        ((float*)outp)[((size_t)b2 * 16384 + nf) * 256 + n] = v;
      }
    }
  }
}

// ---------------------------------------------------------------------------
// Fused per-(batch,head) attention.  L in {64,256}, NB = 32768/L/? (512 or 128).
// Block = 128 threads (4 waves).  K[L,32] and V^T[32,L] staged in LDS (bf16).
// Scores live entirely in WMMA f32 accumulators; softmax reduces across the
// 16-lane halves of the C layout with __shfl_xor(1,2,4,8); P restaged through
// per-wave LDS as bf16 A-fragments for the P x V WMMA.
// ---------------------------------------------------------------------------
template <int L, int NB>
__global__ __launch_bounds__(128) void attn_kernel(const u16* __restrict__ qp,
                                                   const u16* __restrict__ kp,
                                                   const u16* __restrict__ vp,
                                                   u16* __restrict__ op) {
  constexpr int T   = L / 16;   // column tiles of the score matrix
  constexpr int TPW = T / 4;    // row tiles per wave (4 waves)

  __shared__ __align__(16) u16 Ksh[L * 32];      // K row-major [L][32]
  __shared__ __align__(16) u16 Vt[32 * L];       // V transposed [32][L]
  __shared__ __align__(16) u16 Psh[4][16 * L];   // per-wave P staging [16][L]

  const int tid  = threadIdx.x;
  const int wave = tid >> 5;
  const int lane = tid & 31;
  const int hf   = lane >> 4;
  const int l15  = lane & 15;
  const int nb   = blockIdx.x >> 3;
  const int head = blockIdx.x & 7;
  const size_t colBase = (size_t)head * 32;

  for (int idx = tid; idx < L * 32; idx += 128) {
    int l = idx >> 5, dd = idx & 31;
    size_t g = ((size_t)(l * NB + nb)) * 256 + colBase + dd;
    Ksh[idx]        = kp[g];
    Vt[dd * L + l]  = vp[g];
  }
  __syncthreads();

  const float scale = 0.17677669529663687f;      // 1/sqrt(32)

  for (int rt = 0; rt < TPW; ++rt) {
    const int tile = wave * TPW + rt;
    const int qrow = tile * 16 + l15;

    // --- Q A-fragment (16x32, one WMMA K-depth) from global ---
    Frag a;
    const u16* qptr = qp + ((size_t)(qrow * NB + nb)) * 256 + colBase + (hf ? 8 : 0);
    a.q[0] = *(const u32x4*)qptr;
    a.q[1] = *(const u32x4*)(qptr + 16);

    // --- S = Q K^T : T accumulators of 16x16 f32 ---
    v8f accS[T];
#pragma unroll
    for (int t = 0; t < T; ++t) {
      Frag b;
      const u16* kptr = &Ksh[(t * 16 + l15) * 32 + (hf ? 8 : 0)];
      b.q[0] = *(const u32x4*)kptr;
      b.q[1] = *(const u32x4*)(kptr + 16);
      v8f c;
      for (int r = 0; r < 8; ++r) c[r] = 0.f;
      accS[t] = wmma_bf16(a, b, c);
    }

    // --- softmax over the L columns (per C-layout row r / r+8 split) ---
    float inv[8];
#pragma unroll
    for (int r = 0; r < 8; ++r) {
      float m = accS[0][r];
      for (int t = 1; t < T; ++t) m = fmaxf(m, accS[t][r]);
      m = fmaxf(m, __shfl_xor(m, 1, 32));
      m = fmaxf(m, __shfl_xor(m, 2, 32));
      m = fmaxf(m, __shfl_xor(m, 4, 32));
      m = fmaxf(m, __shfl_xor(m, 8, 32));
      float s = 0.f;
      for (int t = 0; t < T; ++t) {
        float p = __expf(scale * (accS[t][r] - m));
        accS[t][r] = p;
        s += p;
      }
      s += __shfl_xor(s, 1, 32);
      s += __shfl_xor(s, 2, 32);
      s += __shfl_xor(s, 4, 32);
      s += __shfl_xor(s, 8, 32);
      inv[r] = 1.0f / s;
    }

    // --- restage P (bf16) through per-wave LDS in row-major [16][L] ---
#pragma unroll
    for (int t = 0; t < T; ++t)
      for (int r = 0; r < 8; ++r) {
        int row = r + hf * 8;
        Psh[wave][row * L + t * 16 + l15] = f2bf(accS[t][r] * inv[r]);
      }

    // --- O = P x V : 2 output tiles (d=32), K-loop over L in steps of 32 ---
    v8f accO[2];
    for (int nt = 0; nt < 2; ++nt)
      for (int r = 0; r < 8; ++r) accO[nt][r] = 0.f;
#pragma unroll
    for (int kt = 0; kt < L / 32; ++kt) {
      Frag pa;
      const u16* pptr = &Psh[wave][l15 * L + kt * 32 + (hf ? 8 : 0)];
      pa.q[0] = *(const u32x4*)pptr;
      pa.q[1] = *(const u32x4*)(pptr + 16);
#pragma unroll
      for (int nt = 0; nt < 2; ++nt) {
        Frag vb;   // B^T row dd = Vt[dd][.] contiguous
        const u16* vptr = &Vt[(nt * 16 + l15) * L + kt * 32 + (hf ? 8 : 0)];
        vb.q[0] = *(const u32x4*)vptr;
        vb.q[1] = *(const u32x4*)(vptr + 16);
        accO[nt] = wmma_bf16(pa, vb, accO[nt]);
      }
    }

    // --- store O tile to [L,NB,C] bf16 ---
#pragma unroll
    for (int nt = 0; nt < 2; ++nt)
      for (int r = 0; r < 8; ++r) {
        int row = tile * 16 + r + hf * 8;
        size_t g = ((size_t)(row * NB + nb)) * 256 + colBase + nt * 16 + l15;
        op[g] = f2bf(accO[nt][r]);
      }
  }
}

// ---------------------------------------------------------------------------
// Host-side orchestration.
// ---------------------------------------------------------------------------
extern "C" void kernel_launch(void* const* d_in, const int* in_sizes, int n_in,
                              void* d_out, int out_size, void* d_ws, size_t ws_size,
                              hipStream_t stream) {
  const float* q   = (const float*)d_in[0];
  const float* k   = (const float*)d_in[1];
  const float* v   = (const float*)d_in[2];
  const float* pos = (const float*)d_in[3];
  const float* wiL = (const float*)d_in[4];
  const float* biL = (const float*)d_in[5];
  const float* woL = (const float*)d_in[6];
  const float* boL = (const float*)d_in[7];
  const float* wiS = (const float*)d_in[8];
  const float* biS = (const float*)d_in[9];
  const float* woS = (const float*)d_in[10];
  const float* boS = (const float*)d_in[11];

  char* ws = (char*)d_ws;
  // bf16 weights at the front of workspace
  u16* wiLb = (u16*)ws;                  // 768*256
  u16* woLb = wiLb + 196608;             // 256*256
  u16* wiSb = woLb + 65536;              // 768*256
  u16* woSb = wiSb + 196608;             // 256*256
  char* slots = ws + (size_t)1048576;
  const size_t SLOT = (size_t)16777216;  // 32768*256 bf16
  auto slot = [&](int i) { return (u16*)(slots + (size_t)i * SLOT); };

  u16* xq1 = slot(0);  u16* xk1 = slot(1);  u16* xv1 = slot(2);
  u16* qp1 = slot(3);  u16* kp1 = slot(4);  u16* vp1 = slot(5);

  const int EW_BLOCKS = (int)(((size_t)32768 * 256 + 255) / 256);
  dim3 gemm_grid(512, 4);

  // weights f32 -> bf16
  cvt_kernel<<<(196608 + 255) / 256, 256, 0, stream>>>(wiL, wiLb, 196608);
  cvt_kernel<<<(65536  + 255) / 256, 256, 0, stream>>>(woL, woLb, 65536);
  cvt_kernel<<<(196608 + 255) / 256, 256, 0, stream>>>(wiS, wiSb, 196608);
  cvt_kernel<<<(65536  + 255) / 256, 256, 0, stream>>>(woS, woSb, 65536);

  // ---- stage 1 (long range, L=64, NB=512) ----
  pack1_kernel<<<EW_BLOCKS, 256, 0, stream>>>(q, k, v, pos, xq1, xk1, xv1);
  gemm_kernel<<<gemm_grid, 128, 0, stream>>>(xq1, wiLb,          biL,       (void*)qp1, 0);
  gemm_kernel<<<gemm_grid, 128, 0, stream>>>(xk1, wiLb + 65536,  biL + 256, (void*)kp1, 0);
  gemm_kernel<<<gemm_grid, 128, 0, stream>>>(xv1, wiLb + 131072, biL + 512, (void*)vp1, 0);

  u16* o1 = slot(0);                                    // xq1 dead
  attn_kernel<64, 512><<<512 * 8, 128, 0, stream>>>(qp1, kp1, vp1, o1);

  float* v1 = (float*)slot(1);                          // spans slots 1..2 (f32)
  gemm_kernel<<<gemm_grid, 128, 0, stream>>>(o1, woLb, boL, (void*)v1, 1);

  // ---- stage 2 (short range, L=256, NB=128) ----
  u16* xqk2 = slot(3);  u16* xv2 = slot(4);             // qp1/kp1 dead
  pack2_kernel<<<EW_BLOCKS, 256, 0, stream>>>(v1, pos, xqk2, xv2);

  u16* qp2 = slot(5);  u16* kp2 = slot(6);  u16* vp2 = slot(7);
  gemm_kernel<<<gemm_grid, 128, 0, stream>>>(xqk2, wiSb,          biS,       (void*)qp2, 0);
  gemm_kernel<<<gemm_grid, 128, 0, stream>>>(xqk2, wiSb + 65536,  biS + 256, (void*)kp2, 0);
  gemm_kernel<<<gemm_grid, 128, 0, stream>>>(xv2,  wiSb + 131072, biS + 512, (void*)vp2, 0);

  u16* o2 = slot(3);                                    // xqk2 dead after gemms
  attn_kernel<256, 128><<<128 * 8, 128, 0, stream>>>(qp2, kp2, vp2, o2);

  // final out-projection scatters straight into the permuted [B,N,C] output
  gemm_kernel<<<gemm_grid, 128, 0, stream>>>(o2, woSb, boS, d_out, 2);

  (void)in_sizes; (void)n_in; (void)out_size; (void)ws_size;
}
